// Multi_JointContextTransfer_14396730376603
// MI455X (gfx1250) — matrix operand order repaired and checked by reading, not compiled, not run on hardware
//
#include <hip/hip_runtime.h>
#include <hip/hip_bf16.h>

typedef __attribute__((ext_vector_type(16))) _Float16 v16h;
typedef __attribute__((ext_vector_type(8)))  _Float16 v8h;
typedef __attribute__((ext_vector_type(8)))  float    v8f;
typedef __attribute__((ext_vector_type(2)))  float    v2f;
typedef int v4i32 __attribute__((vector_size(4 * sizeof(int))));

constexpr int kC  = 128;
constexpr int kH  = 128;
constexpr int kW  = 128;
constexpr int kHW = kH * kW;          // 16384
constexpr int kN  = 8;                // batch*views
constexpr int kNHW = kN * kHW;        // 131072

// ---------------- workspace layout (in floats) ----------------
constexpr size_t SZBUF   = (size_t)kN * kC * kHW;     // 16,777,216
constexpr size_t OFF_A   = 0;
constexpr size_t OFF_B   = SZBUF;
constexpr size_t OFF_Cb  = 2 * SZBUF;
constexpr size_t OFF_K   = 3 * SZBUF;                 // 8*32*16384 = 4,194,304
constexpr size_t OFF_Q   = OFF_K + (size_t)kN * 32 * kHW;
constexpr size_t OFF_V   = OFF_Q + (size_t)kN * 32 * kHW;
constexpr size_t OFF_CP  = OFF_V + (size_t)kN * 64 * kHW;   // ctx partials 16*64*512
constexpr size_t OFF_CTX = OFF_CP + (size_t)16 * 64 * 512;  // 8192
constexpr size_t OFF_WPK = OFF_CTX + 8192;                  // f16 weights: 147456 halves
constexpr size_t OFF_X16 = OFF_WPK + 147456 / 2 + 32;       // f16 NHWC activations
constexpr size_t WS_FLOATS = OFF_X16 + SZBUF / 2 + 64;

// ---------------- CDNA5 async global->LDS (guarded; falls back to reg copy) ----
#if defined(__AMDGCN__) && __has_builtin(__builtin_amdgcn_global_load_async_to_lds_b128) && \
    __has_builtin(__builtin_amdgcn_s_wait_asynccnt)
#define USE_ASYNC_LDS 1
#else
#define USE_ASYNC_LDS 0
#endif

__device__ __forceinline__ void copy16_g2l(const _Float16* __restrict__ g,
                                           _Float16* __restrict__ l) {
#if USE_ASYNC_LDS
  __builtin_amdgcn_global_load_async_to_lds_b128(
      (__attribute__((address_space(1))) v4i32*)g,
      (__attribute__((address_space(3))) v4i32*)l, 0, 0);
#else
  *(v8h*)l = *(const v8h*)g;
#endif
}

__device__ __forceinline__ void wait_async_lds() {
#if USE_ASYNC_LDS
  __builtin_amdgcn_s_wait_asynccnt(0);
#endif
}

// ---------------- NCHW f32 -> NHWC f16 activation convert ----------------
__global__ __launch_bounds__(256)
void nchw_to_nhwc_f16_kernel(const float* __restrict__ in, _Float16* __restrict__ out) {
  const int t = blockIdx.x * blockDim.x + threadIdx.x;   // 16,777,216
  if (t >= kN * kC * kHW) return;
  __builtin_prefetch(in + t + 65536, 0, 0);              // global_prefetch_b8
  const int x = t & 127;
  const int y = (t >> 7) & 127;
  const int c = (t >> 14) & 127;
  const int n = t >> 21;
  out[(((size_t)n * kH + y) * kW + x) * kC + c] = (_Float16)in[t];
}

// ---------------- weight pack: OIHW f32 -> [tap][co][ci] f16 ----------------
__global__ void pack_w3_kernel(const float* __restrict__ w, _Float16* __restrict__ wpk) {
  const int t = blockIdx.x * blockDim.x + threadIdx.x;
  if (t >= kC * kC * 9) return;
  const int kw = t % 3, kh = (t / 3) % 3, ci = (t / 9) % kC, co = t / (9 * kC);
  wpk[((size_t)(kh * 3 + kw) * kC + co) * kC + ci] = (_Float16)w[t];
}

// ---------------- 3x3 conv, implicit GEMM, f16 WMMA, fused BN/res/relu ----------------
// input: f16 NHWC (pre-converted); residuals/output: f32 NCHW
__global__ __launch_bounds__(128)
void conv3x3_bn_kernel(const _Float16* __restrict__ in16, const _Float16* __restrict__ wpk,
                       const float* __restrict__ gg, const float* __restrict__ bb,
                       const float* __restrict__ mm, const float* __restrict__ vv,
                       const float* __restrict__ res1, const float* __restrict__ res2,
                       float* __restrict__ out) {
  __shared__ __align__(32) _Float16 smem[3 * 66 * 128];   // [row][col][ci], ~50KB
  const int tid  = threadIdx.x;
  const int blk  = blockIdx.x;
  const int xblk = blk & 1;
  const int y    = (blk >> 1) & 127;
  const int n    = blk >> 8;
  const int x0   = xblk * 64;

  // stage 3x66x128 halo tile: async b128 copies (NHWC is contiguous in ci),
  // 16 lanes cooperate per (row,col) site; halo sites -> zero ds_store.
  {
    const int sl = tid & 15;                 // lane within site: 8 ci-halves
    const v8h zero = {};
    for (int site = tid >> 4; site < 3 * 66; site += 8) {
      const int row = site / 66, col = site % 66;
      const int gy = y - 1 + row, gx = x0 - 1 + col;
      _Float16* ldst = smem + ((row * 66 + col) * 128 + sl * 8);
      if (gy >= 0 && gy < kH && gx >= 0 && gx < kW) {
        const _Float16* gsrc = in16 + (((size_t)n * kH + gy) * kW + gx) * kC + sl * 8;
        copy16_g2l(gsrc, ldst);
      } else {
        *(v8h*)ldst = zero;
      }
    }
  }
  wait_async_lds();
  __syncthreads();

  const int wave = tid >> 5, lane = tid & 31, ln = lane & 15, hi = lane >> 4;
  const int mbase = wave * 32;

  v8f acc[2][4];
  #pragma unroll
  for (int mt = 0; mt < 2; ++mt)
    #pragma unroll
    for (int nt = 0; nt < 4; ++nt) acc[mt][nt] = v8f{};

  for (int tap = 0; tap < 9; ++tap) {
    const int kh = tap / 3, kwp = tap % 3;
    #pragma unroll
    for (int cc = 0; cc < kC; cc += 32) {
      v16h afrag[2];
      #pragma unroll
      for (int mt = 0; mt < 2; ++mt) {
        const _Float16* wb = wpk + ((size_t)(tap * kC + (mbase + mt * 16 + ln))) * kC + cc;
        const v8h alo = *(const v8h*)(wb + hi * 16);
        const v8h ahi = *(const v8h*)(wb + hi * 16 + 8);
        v16h a;
        #pragma unroll
        for (int i = 0; i < 8; ++i) { a[i] = alo[i]; a[8 + i] = ahi[i]; }
        afrag[mt] = a;
      }
      #pragma unroll
      for (int nt = 0; nt < 4; ++nt) {
        const int col = nt * 16 + ln + kwp;
        const v16h bfrag = *(const v16h*)(smem + ((kh * 66 + col) * 128 + cc + hi * 16));
        acc[0][nt] = __builtin_amdgcn_wmma_f32_16x16x32_f16(
            false, afrag[0], false, bfrag, (short)0, acc[0][nt], false, false);
        acc[1][nt] = __builtin_amdgcn_wmma_f32_16x16x32_f16(
            false, afrag[1], false, bfrag, (short)0, acc[1][nt], false, false);
      }
    }
  }

  #pragma unroll
  for (int mt = 0; mt < 2; ++mt) {
    float sC[8], tC[8];
    #pragma unroll
    for (int j = 0; j < 8; ++j) {
      const int co = mbase + mt * 16 + j + hi * 8;
      const float sc = gg[co] * rsqrtf(vv[co] + 1e-5f);
      sC[j] = sc; tC[j] = bb[co] - mm[co] * sc;
    }
    #pragma unroll
    for (int nt = 0; nt < 4; ++nt) {
      const int x = x0 + nt * 16 + ln;
      #pragma unroll
      for (int j = 0; j < 8; ++j) {
        const int co = mbase + mt * 16 + j + hi * 8;
        const size_t idx = (((size_t)n * kC + co) * kH + y) * kW + x;
        float val = acc[mt][nt][j] * sC[j] + tC[j];
        if (res1) val += res1[idx];
        val = fmaxf(val, 0.f);
        if (res2) val += res2[idx];
        out[idx] = val;
      }
    }
  }
}

// ---------------- 1x1 conv GEMM (f32 WMMA 16x16x4), optional concat / bias / BN / relu ----
template <int MT>
__global__ __launch_bounds__(128)
void gemm1x1_kernel(const float* __restrict__ in0, int K0,
                    const float* __restrict__ in1, int K1,
                    const float* __restrict__ Wm,
                    const float* __restrict__ bias,
                    const float* __restrict__ gg, const float* __restrict__ bb,
                    const float* __restrict__ mm, const float* __restrict__ vv,
                    int relu, float* __restrict__ out) {
  const int tid = threadIdx.x;
  const int wave = tid >> 5, lane = tid & 31, ln = lane & 15, hi = lane >> 4;
  const int hw0  = blockIdx.x * 64 + wave * 16;
  const int n    = hw0 >> 14;
  const int hwin = hw0 & (kHW - 1);
  const int K    = K0 + K1;
  constexpr int M = MT * 16;

  v8f acc[MT];
  #pragma unroll
  for (int mt = 0; mt < MT; ++mt) acc[mt] = v8f{};

  for (int k = 0; k < K; k += 4) {
    v2f bf;
    #pragma unroll
    for (int p = 0; p < 2; ++p) {
      const int kk = k + 2 * hi + p;
      const float* src = (kk < K0) ? (in0 + ((size_t)n * K0 + kk) * kHW)
                                   : (in1 + ((size_t)n * K1 + (kk - K0)) * kHW);
      bf[p] = src[hwin + ln];
    }
    #pragma unroll
    for (int mt = 0; mt < MT; ++mt) {
      const v2f af = *(const v2f*)(Wm + (size_t)(mt * 16 + ln) * K + k + 2 * hi);
      acc[mt] = __builtin_amdgcn_wmma_f32_16x16x4_f32(
          false, af, false, bf, (short)0, acc[mt], false, false);
    }
  }
  #pragma unroll
  for (int mt = 0; mt < MT; ++mt) {
    #pragma unroll
    for (int j = 0; j < 8; ++j) {
      const int co = mt * 16 + j + hi * 8;
      float val = acc[mt][j];
      if (bias) val += bias[co];
      if (gg) { const float sc = gg[co] * rsqrtf(vv[co] + 1e-5f);
                val = val * sc + (bb[co] - mm[co] * sc); }
      if (relu) val = fmaxf(val, 0.f);
      out[((size_t)n * M + co) * kHW + hwin + ln] = val;
    }
  }
}

// ---------------- cross-view aggregation: mean of other views ----------------
__global__ void agg_kernel(const float* __restrict__ r, float* __restrict__ agg) {
  const int t = blockIdx.x * blockDim.x + threadIdx.x;   // 2*128*16384
  if (t >= 2 * kC * kHW) return;
  const int hw = t & (kHW - 1);
  const int c  = (t >> 14) & 127;
  const int b  = t >> 21;
  const size_t stride = (size_t)kC * kHW;
  const size_t base = ((size_t)(b * 4) * kC + c) * kHW + hw;
  const float v0 = r[base], v1 = r[base + stride], v2 = r[base + 2 * stride], v3 = r[base + 3 * stride];
  const float s = v0 + v1 + v2 + v3;
  const float inv3 = (1.f / 3.f);
  agg[base]              = (s - v0) * inv3;
  agg[base + stride]     = (s - v1) * inv3;
  agg[base + 2 * stride] = (s - v2) * inv3;
  agg[base + 3 * stride] = (s - v3) * inv3;
}

// ---------------- softmax over spatial axis (k) ----------------
__global__ __launch_bounds__(256)
void softmax_spatial_kernel(float* __restrict__ kb) {
  __shared__ float red[256];
  float* p = kb + (size_t)blockIdx.x * kHW;   // 256 rows = [n][32]
  const int tid = threadIdx.x;
  float m = -1e30f;
  for (int i = tid; i < kHW; i += 256) m = fmaxf(m, p[i]);
  red[tid] = m; __syncthreads();
  for (int s = 128; s > 0; s >>= 1) { if (tid < s) red[tid] = fmaxf(red[tid], red[tid + s]); __syncthreads(); }
  m = red[0]; __syncthreads();
  float sum = 0.f;
  for (int i = tid; i < kHW; i += 256) sum += __expf(p[i] - m);
  red[tid] = sum; __syncthreads();
  for (int s = 128; s > 0; s >>= 1) { if (tid < s) red[tid] += red[tid + s]; __syncthreads(); }
  const float inv = 1.f / red[0];
  for (int i = tid; i < kHW; i += 256) p[i] = __expf(p[i] - m) * inv;
}

// ---------------- softmax over key-dim axis (q), fused *SCALE ----------------
__global__ __launch_bounds__(256)
void softmax_channel_kernel(float* __restrict__ qb) {
  const int t = blockIdx.x * blockDim.x + threadIdx.x;   // 8*2*16384
  if (t >= kN * 2 * kHW) return;
  const int hw = t & (kHW - 1);
  const int h  = (t >> 14) & 1;
  const int n  = t >> 15;
  float* base = qb + ((size_t)n * 32 + h * 16) * kHW + hw;
  float vals[16]; float m = -1e30f;
  #pragma unroll
  for (int d = 0; d < 16; ++d) { vals[d] = base[(size_t)d * kHW] * 0.25f; m = fmaxf(m, vals[d]); }
  float s = 0.f;
  #pragma unroll
  for (int d = 0; d < 16; ++d) { vals[d] = __expf(vals[d] - m); s += vals[d]; }
  const float inv = 1.f / s;
  #pragma unroll
  for (int d = 0; d < 16; ++d) base[(size_t)d * kHW] = vals[d] * inv;
}

// ---------------- ctx = k @ v^T per (b,h), split-K partials (deterministic) ----------------
__global__ __launch_bounds__(32)
void ctx_partial_kernel(const float* __restrict__ kbuf, const float* __restrict__ vbuf,
                        float* __restrict__ part) {
  const int lane = threadIdx.x, ln = lane & 15, hi = lane >> 4;
  const int bh = blockIdx.x >> 6;     // 0..15
  const int kc = blockIdx.x & 63;     // K chunk
  const int n = bh >> 1, h = bh & 1;
  const int hwb = kc * 256;
  const float* kp  = kbuf + ((size_t)n * 32 + h * 16 + ln) * kHW;
  const float* vp0 = vbuf + ((size_t)n * 64 + h * 32 + ln) * kHW;
  const float* vp1 = vp0 + (size_t)16 * kHW;
  v8f a0 = v8f{}, a1 = v8f{};
  for (int s = 0; s < 64; ++s) {
    const int hw = hwb + s * 4 + 2 * hi;
    const v2f af = *(const v2f*)(kp + hw);
    const v2f b0 = *(const v2f*)(vp0 + hw);
    const v2f b1 = *(const v2f*)(vp1 + hw);
    a0 = __builtin_amdgcn_wmma_f32_16x16x4_f32(false, af, false, b0, (short)0, a0, false, false);
    a1 = __builtin_amdgcn_wmma_f32_16x16x4_f32(false, af, false, b1, (short)0, a1, false, false);
  }
  float* dst = part + ((size_t)(bh * 64 + kc)) * 512;
  #pragma unroll
  for (int j = 0; j < 8; ++j) {
    const int d = j + 8 * hi;
    dst[d * 32 + ln]      = a0[j];
    dst[d * 32 + 16 + ln] = a1[j];
  }
}

__global__ void ctx_reduce_kernel(const float* __restrict__ part, float* __restrict__ ctx) {
  const int t = blockIdx.x * blockDim.x + threadIdx.x;   // 16*512
  if (t >= 16 * 512) return;
  const int bh = t >> 9, de = t & 511;
  float s = 0.f;
  for (int kc = 0; kc < 64; ++kc) s += part[((size_t)(bh * 64 + kc) << 9) + de];
  ctx[t] = s;
}

// ---------------- out[e][hw] = sum_d ctx[d][e] * q[d][hw] ----------------
__global__ __launch_bounds__(128)
void attn_out_kernel(const float* __restrict__ ctx, const float* __restrict__ qbuf,
                     float* __restrict__ outb) {
  const int tid = threadIdx.x;
  const int wave = tid >> 5, lane = tid & 31, ln = lane & 15, hi = lane >> 4;
  const int bh  = blockIdx.x >> 8;
  const int hw0 = (blockIdx.x & 255) * 64 + wave * 16;
  const int n = bh >> 1, h = bh & 1;
  v8f acc[2]; acc[0] = v8f{}; acc[1] = v8f{};
  #pragma unroll
  for (int k = 0; k < 16; k += 4) {
    v2f bf;
    #pragma unroll
    for (int p = 0; p < 2; ++p)
      bf[p] = qbuf[((size_t)n * 32 + h * 16 + k + 2 * hi + p) * kHW + hw0 + ln];
    #pragma unroll
    for (int mt = 0; mt < 2; ++mt) {
      v2f af;
      #pragma unroll
      for (int p = 0; p < 2; ++p)
        af[p] = ctx[((size_t)bh * 16 + k + 2 * hi + p) * 32 + mt * 16 + ln];
      acc[mt] = __builtin_amdgcn_wmma_f32_16x16x4_f32(
          false, af, false, bf, (short)0, acc[mt], false, false);
    }
  }
  #pragma unroll
  for (int mt = 0; mt < 2; ++mt)
    #pragma unroll
    for (int j = 0; j < 8; ++j) {
      const int e = mt * 16 + j + 8 * hi;
      outb[((size_t)n * 64 + h * 32 + e) * kHW + hw0 + ln] = acc[mt][j];
    }
}

// =======================================================================
extern "C" void kernel_launch(void* const* d_in, const int* in_sizes, int n_in,
                              void* d_out, int out_size, void* d_ws, size_t ws_size,
                              hipStream_t stream) {
  (void)in_sizes; (void)n_in; (void)out_size;
  if (ws_size < WS_FLOATS * sizeof(float)) return;

  const float* x = (const float*)d_in[0];
  // flattened param order: rb1, rb2, attn, refine (see setup_inputs dict order)
  const float* rb1w1 = (const float*)d_in[2];
  const float* rb1g1 = (const float*)d_in[3];  const float* rb1b1 = (const float*)d_in[4];
  const float* rb1m1 = (const float*)d_in[5];  const float* rb1v1 = (const float*)d_in[6];
  const float* rb1w2 = (const float*)d_in[7];
  const float* rb1g2 = (const float*)d_in[8];  const float* rb1b2 = (const float*)d_in[9];
  const float* rb1m2 = (const float*)d_in[10]; const float* rb1v2 = (const float*)d_in[11];
  const float* rb2w1 = (const float*)d_in[12];
  const float* rb2g1 = (const float*)d_in[13]; const float* rb2b1 = (const float*)d_in[14];
  const float* rb2m1 = (const float*)d_in[15]; const float* rb2v1 = (const float*)d_in[16];
  const float* rb2w2 = (const float*)d_in[17];
  const float* rb2g2 = (const float*)d_in[18]; const float* rb2b2 = (const float*)d_in[19];
  const float* rb2m2 = (const float*)d_in[20]; const float* rb2v2 = (const float*)d_in[21];
  const float* wk = (const float*)d_in[22]; const float* bk = (const float*)d_in[23];
  const float* wq = (const float*)d_in[24]; const float* bq = (const float*)d_in[25];
  const float* wv = (const float*)d_in[26]; const float* bv = (const float*)d_in[27];
  const float* wr = (const float*)d_in[28]; const float* br = (const float*)d_in[29];
  const float* rfw = (const float*)d_in[30];
  const float* rfg = (const float*)d_in[31]; const float* rfb = (const float*)d_in[32];
  const float* rfm = (const float*)d_in[33]; const float* rfv = (const float*)d_in[34];
  const float* rrw1 = (const float*)d_in[35];
  const float* rrg1 = (const float*)d_in[36]; const float* rrb1 = (const float*)d_in[37];
  const float* rrm1 = (const float*)d_in[38]; const float* rrv1 = (const float*)d_in[39];
  const float* rrw2 = (const float*)d_in[40];
  const float* rrg2 = (const float*)d_in[41]; const float* rrb2 = (const float*)d_in[42];
  const float* rrm2 = (const float*)d_in[43]; const float* rrv2 = (const float*)d_in[44];

  float* ws = (float*)d_ws;
  float* bufA = ws + OFF_A;
  float* bufB = ws + OFF_B;
  float* bufC = ws + OFF_Cb;
  float* kbuf = ws + OFF_K;
  float* qbuf = ws + OFF_Q;
  float* vbuf = ws + OFF_V;
  float* cpart = ws + OFF_CP;
  float* ctxb  = ws + OFF_CTX;
  _Float16* wpk = (_Float16*)(ws + OFF_WPK);
  _Float16* x16 = (_Float16*)(ws + OFF_X16);
  float* outp = (float*)d_out;

  auto conv3 = [&](const float* in, const float* w, const float* g, const float* b,
                   const float* m, const float* v, const float* r1, const float* r2,
                   float* out) {
    nchw_to_nhwc_f16_kernel<<<(kN * kC * kHW) / 256, 256, 0, stream>>>(in, x16);
    pack_w3_kernel<<<(kC * kC * 9 + 255) / 256, 256, 0, stream>>>(w, wpk);
    conv3x3_bn_kernel<<<kN * kH * 2, 128, 0, stream>>>(x16, wpk, g, b, m, v, r1, r2, out);
  };

  // residual block 1: A = relu(bn1(conv(x)));  B = relu(x + bn2(conv(A)))
  conv3(x,    rb1w1, rb1g1, rb1b1, rb1m1, rb1v1, nullptr, nullptr, bufA);
  conv3(bufA, rb1w2, rb1g2, rb1b2, rb1m2, rb1v2, x,       nullptr, bufB);
  // residual block 2: A = ...; C = r = relu(B + bn2(conv(A)))
  conv3(bufB, rb2w1, rb2g1, rb2b1, rb2m1, rb2v1, nullptr, nullptr, bufA);
  conv3(bufA, rb2w2, rb2g2, rb2b2, rb2m2, rb2v2, bufB,    nullptr, bufC);

  // agg (mean of other views): C=r -> A=agg
  agg_kernel<<<(2 * kC * kHW + 255) / 256, 256, 0, stream>>>(bufC, bufA);

  // k,v from agg(A); q from r(C)
  gemm1x1_kernel<2><<<kNHW / 64, 128, 0, stream>>>(bufA, 128, nullptr, 0, wk, bk,
      nullptr, nullptr, nullptr, nullptr, 0, kbuf);
  gemm1x1_kernel<2><<<kNHW / 64, 128, 0, stream>>>(bufC, 128, nullptr, 0, wq, bq,
      nullptr, nullptr, nullptr, nullptr, 0, qbuf);
  gemm1x1_kernel<4><<<kNHW / 64, 128, 0, stream>>>(bufA, 128, nullptr, 0, wv, bv,
      nullptr, nullptr, nullptr, nullptr, 0, vbuf);

  softmax_spatial_kernel<<<kN * 32, 256, 0, stream>>>(kbuf);
  softmax_channel_kernel<<<(kN * 2 * kHW + 255) / 256, 256, 0, stream>>>(qbuf);

  // ctx[b,h,16,32] = k @ v^T  (split-K, deterministic reduction)
  ctx_partial_kernel<<<16 * 64, 32, 0, stream>>>(kbuf, vbuf, cpart);
  ctx_reduce_kernel<<<(16 * 512 + 255) / 256, 256, 0, stream>>>(cpart, ctxb);

  // out = ctx^T @ q  -> overwrite vbuf (v no longer needed)
  attn_out_kernel<<<16 * 256, 128, 0, stream>>>(ctxb, qbuf, vbuf);

  // A = wr @ attn_out + br  (agg no longer needed)
  gemm1x1_kernel<8><<<kNHW / 64, 128, 0, stream>>>(vbuf, 64, nullptr, 0, wr, br,
      nullptr, nullptr, nullptr, nullptr, 0, bufA);

  // h = relu(bn(W_cat @ concat(A, r)))  -> B (r1 no longer needed)
  gemm1x1_kernel<8><<<kNHW / 64, 128, 0, stream>>>(bufA, 128, bufC, 128, rfw, nullptr,
      rfg, rfb, rfm, rfv, 1, bufB);

  // refine residual block; final conv fuses out = x + relu(h + bn(conv))
  conv3(bufB, rrw1, rrg1, rrb1, rrm1, rrv1, nullptr, nullptr, bufC);
  conv3(bufC, rrw2, rrg2, rrb2, rrm2, rrv2, bufB,    (const float*)x, outp);
}
